// m_btabl_29360396436127
// MI455X (gfx1250) — matrix operand order repaired and checked by reading, not compiled, not run on hardware
//
#include <hip/hip_runtime.h>
#include <hip/hip_bf16.h>
#include <math.h>

// ---------------------------------------------------------------------------
// m_btabl forward for MI455X (gfx1250), bf16 WMMA, fully fused per 8 items.
//   Stage 1a: hpre[e,(b,t)]   = W1a @ x          (WMMA, M=120 K=40 N=80)
//   Stage 1b: h2[(e,b),tau]   = relu(hpre@W2a+B) (WMMA, data-as-A, K=10)
//   Stage 2 : X[c,(b,tau)]    = W1b @ h2         (WMMA, M=3 K=120 N=40)
//   Tail    : 5-wide attention softmax + 3-way softmax (VALU, 8 lanes)
// Roofline: 211 MB HBM traffic -> ~9us floor @23.3TB/s; matrix work mapped
// to v_wmma_f32_16x16x32_bf16 (~41 GFLOP issued) stays under that floor.
// ---------------------------------------------------------------------------

typedef __attribute__((ext_vector_type(16))) __bf16       v16bf;
typedef __attribute__((ext_vector_type(8)))  float        v8f;
typedef __attribute__((ext_vector_type(8)))  unsigned int v8u;

// workspace layout (dwords)
#define WS_P    0                  // [0]=lam, [1..25]=Wm, [26..30]=W2b*s5
#define WS_A1   64                 // 4096 dw: A-frags of W1a  [8 mt][2 ks][32][8]
#define WS_BW2  (64 + 4096)        // 256 dw : B-frag  of W2a  [32][8]
#define WS_A2   (64 + 4096 + 256)  // 1024 dw: A-frags of W1b  [4 ks][32][8]

__device__ __forceinline__ unsigned pkbf(float a, float b) {
  // pack two floats as bf16 pair (round-to-nearest-even)
  unsigned ua = __float_as_uint(a);
  unsigned ub = __float_as_uint(b);
  ua = (ua + 0x7FFFu + ((ua >> 16) & 1u)) >> 16;
  ub = (ub + 0x7FFFu + ((ub >> 16) & 1u)) >> 16;
  return (ua & 0xFFFFu) | (ub << 16);
}

// A-operand (MxK, 16x32 bf16) K index for (vgpr j, lane): ISA 7.12.2 table
__device__ __forceinline__ int kkA(int j, int lane) {
  return ((j < 4) ? 0 : 16) + ((lane < 16) ? 0 : 8) + ((j & 3) << 1);
}
// B-operand (KxN, 32x16 bf16) K index: same table order as A
__device__ __forceinline__ int kkB(int j, int lane) {
  return ((j < 4) ? 0 : 16) + ((lane < 16) ? 0 : 8) + ((j & 3) << 1);
}

__device__ __forceinline__ v8f wmma_bf16(v8u a, v8u b, v8f c) {
  return __builtin_amdgcn_wmma_f32_16x16x32_bf16(
      false, __builtin_bit_cast(v16bf, a),
      false, __builtin_bit_cast(v16bf, b),
      (short)0, c, false, false);
}

// ---------------------------------------------------------------------------
// Prep: max-norm scales, Wm / lam params, constant WMMA fragment images.
// ---------------------------------------------------------------------------
__global__ void __launch_bounds__(256) m_btabl_prep(
    const float* __restrict__ W1,  const float* __restrict__ W2,
    const float* __restrict__ TW1, const float* __restrict__ TW,
    const float* __restrict__ TW2, const float* __restrict__ l,
    float* __restrict__ wsf) {
  __shared__ float red[256];
  const int tid = threadIdx.x;

  const float* ptrs[5] = {W1, W2, TW1, TW, TW2};
  const int    ns[5]   = {4800, 50, 360, 25, 5};
  float sc[5];
  for (int q = 0; q < 5; ++q) {
    float s = 0.f;
    const float* p = ptrs[q];
    for (int i = tid; i < ns[q]; i += 256) s += p[i] * p[i];
    red[tid] = s;
    __syncthreads();
    for (int off = 128; off > 0; off >>= 1) {
      if (tid < off) red[tid] += red[tid + off];
      __syncthreads();
    }
    float nrm = sqrtf(red[0]);
    sc[q] = (nrm > 10.f) ? (10.f / (1e-8f + nrm)) : 1.f;
    __syncthreads();
  }

  unsigned* wsu = (unsigned*)wsf;
  if (tid == 0) {
    float lam = fminf(fmaxf(l[0], 0.f), 1.f);
    wsf[WS_P + 0] = lam;
  }
  if (tid < 25) { // Wm = TW*s4 off-diag, 1/5 on diag
    int t = tid / 5, t2 = tid % 5;
    wsf[WS_P + 1 + tid] = (t == t2) ? 0.2f : TW[tid] * sc[3];
  }
  if (tid < 5) wsf[WS_P + 26 + tid] = TW2[tid] * sc[4];

  // A1: W1a fragments, M = e (pad 128), K = d (pad 64)
  for (int idx = tid; idx < 4096; idx += 256) {
    int j = idx & 7, lane = (idx >> 3) & 31, f = idx >> 8;
    int mt = f >> 1, ks = f & 1;
    int m = mt * 16 + (lane & 15);
    int k = ks * 32 + kkA(j, lane);
    float f0 = 0.f, f1 = 0.f;
    if (m < 120 && k < 40) {          // k even -> k+1 also < 40
      f0 = W1[m * 40 + k] * sc[0];
      f1 = W1[m * 40 + k + 1] * sc[0];
    }
    wsu[WS_A1 + idx] = pkbf(f0, f1);
  }
  // BW2: W2a as B-operand, K = t (pad 32), N = tau (pad 16)
  if (tid < 256) {
    int j = tid & 7, lane = tid >> 3;
    int tau = lane & 15;
    int t = kkB(j, lane);
    float f0 = (tau < 5 && t < 10)     ? W2[t * 5 + tau] * sc[1] : 0.f;
    float f1 = (tau < 5 && t + 1 < 10) ? W2[(t + 1) * 5 + tau] * sc[1] : 0.f;
    wsu[WS_BW2 + tid] = pkbf(f0, f1);
  }
  // A2: W1b fragments, M = c (pad 16), K = e (pad 128)
  for (int idx = tid; idx < 1024; idx += 256) {
    int j = idx & 7, lane = (idx >> 3) & 31, ks = idx >> 8;
    int c = lane & 15;
    int e = ks * 32 + kkA(j, lane);
    float f0 = (c < 3 && e < 120) ? TW1[c * 120 + e] * sc[2] : 0.f;
    float f1 = (c < 3 && e < 120) ? TW1[c * 120 + e + 1] * sc[2] : 0.f;
    wsu[WS_A2 + idx] = pkbf(f0, f1);
  }
}

// ---------------------------------------------------------------------------
// Main fused kernel: one block = 8 batch items, 8 waves (wave32).
// LDS (62.5 KB), phase-aliased:
//   [    0,12800) xs   : staged x, f32             (dead after P1)
//   [12800,23040) B1   : x B-fragments             (dead after P2)
//   [    0,19200) h2   : relu output, f32          (born P3, aliases xs/B1)
//   [19200,19840) Xs   : stage-2 output / P3 dump  (born P4)
//   [23040,64000) hpre : stage-1a output, 128 rows (pad kills store guards)
// ---------------------------------------------------------------------------
__global__ void __launch_bounds__(256) m_btabl_main(
    const float* __restrict__ x, const float* __restrict__ BLB,
    const float* __restrict__ TB, const float* __restrict__ wsf,
    float* __restrict__ out, int nB) {
  __shared__ __align__(64) char smem[64000];
  float*    xs   = (float*)smem;
  unsigned* B1   = (unsigned*)(smem + 12800);
  float*    h2   = (float*)smem;
  float*    Xs   = (float*)(smem + 19200);
  float*    hpre = (float*)(smem + 23040);   // 128 x 80 f32 (rows 120..127 pad)

  const int tid  = threadIdx.x;
  const int lane = tid & 31;
  const int wid  = tid >> 5;
  const long b0  = (long)blockIdx.x * 8;
  const unsigned* wsu = (const unsigned*)wsf;

  // ---- P0: stage x (coalesced, 128-bit) ----------------------------------
  {
    const float4* xb4 = (const float4*)(x + b0 * 400);
    const long lim4 = ((long)nB * 400 - b0 * 400) >> 2;  // whole float4s valid
    float4* xs4 = (float4*)xs;
    for (int i = tid; i < 800; i += 256) {
      float4 v = make_float4(0.f, 0.f, 0.f, 0.f);
      if ((long)i < lim4) v = xb4[i];
      xs4[i] = v;
    }
  }
  __syncthreads();

  // ---- P1: build x B-fragments (K = d, N = item*10+t) --------------------
  for (int idx = tid; idx < 2560; idx += 256) {
    int j = idx & 7, ln = (idx >> 3) & 31, f = idx >> 8;
    int ks = f / 5, nt = f % 5;
    int ncol = nt * 16 + (ln & 15);          // 0..79
    int k = ks * 32 + kkB(j, ln);            // d (even)
    int ni = ncol / 10, t = ncol % 10;
    float f0 = 0.f, f1 = 0.f;
    if (k < 40) {
      f0 = xs[ni * 400 + k * 10 + t];
      f1 = xs[ni * 400 + (k + 1) * 10 + t];
    }
    B1[idx] = pkbf(f0, f1);
  }
  __syncthreads();

  // ---- P2: GEMM1a  hpre = W1a @ x   (40 tiles, 5 per wave) ---------------
  {
    const v8u* A1v = (const v8u*)(wsu + WS_A1);
    const v8u* B1v = (const v8u*)B1;
    for (int tile = wid; tile < 40; tile += 8) {
      int mt = tile / 5, nt = tile % 5;
      v8f acc;
#pragma unroll
      for (int i = 0; i < 8; ++i) acc[i] = 0.f;
#pragma unroll
      for (int ks = 0; ks < 2; ++ks) {
        v8u a = A1v[(mt * 2 + ks) * 32 + lane];
        v8u b = B1v[(ks * 5 + nt) * 32 + lane];
        acc = wmma_bf16(a, b, acc);
      }
      int ncol = nt * 16 + (lane & 15);
      int hi = lane >> 4;
      float* hp = hpre + (mt * 16 + hi * 8) * 80 + ncol;  // rows < 128 always
#pragma unroll
      for (int r = 0; r < 8; ++r) hp[r * 80] = acc[r];    // unconditional
    }
  }
  __syncthreads();

  // ---- P3: GEMM1b  h2 = relu(hpre @ W2a + B)  (60 tiles, data-as-A) ------
  {
    v8u bw2 = ((const v8u*)(wsu + WS_BW2))[lane];
    float* dump3 = Xs + lane;                // dead LDS during P3
    for (int tile = wid; tile < 60; tile += 8) {
      int rho = tile * 16 + (lane & 15);     // row = e*8 + item
      int e = rho >> 3, ni = rho & 7;
      const float* hp = hpre + e * 80 + ni * 10;
      v8u a;
#pragma unroll
      for (int j = 0; j < 8; ++j) {
        int kk = kkA(j, lane);               // t (even)
        float f0 = (kk < 10) ? hp[kk] : 0.f;
        float f1 = (kk + 1 < 10) ? hp[kk + 1] : 0.f;
        a[j] = pkbf(f0, f1);
      }
      v8f acc;
#pragma unroll
      for (int i = 0; i < 8; ++i) acc[i] = 0.f;
      acc = wmma_bf16(a, bw2, acc);
      int tau = lane & 15;
      int btau = (tau < 5) ? tau : 0;        // keep BLB reads in range
      int hi = lane >> 4;
#pragma unroll
      for (int r = 0; r < 8; ++r) {
        int rr = tile * 16 + r + hi * 8;
        int ee = rr >> 3, nn = rr & 7;
        float v = fmaxf(acc[r] + BLB[ee * 5 + btau], 0.f);
        float* dst = (tau < 5) ? (h2 + ee * 40 + nn * 5 + tau) : dump3;
        *dst = v;                            // unconditional, addr-selected
      }
    }
  }
  __syncthreads();

  // ---- P4: GEMM2  X = W1b @ h2   (3 N-tiles -> waves 0..2) ---------------
  if (wid < 3) {
    int col2 = wid * 16 + (lane & 15);       // item*5 + tau, 0..47 (40 live)
    v8f acc;
#pragma unroll
    for (int i = 0; i < 8; ++i) acc[i] = 0.f;
    const v8u* A2v = (const v8u*)(wsu + WS_A2);
#pragma unroll
    for (int ks = 0; ks < 4; ++ks) {
      v8u a = A2v[ks * 32 + lane];
      v8u b;
#pragma unroll
      for (int j = 0; j < 8; ++j) {
        int e = ks * 32 + kkB(j, lane);      // e (even)
        float f0 = (col2 < 40 && e < 120) ? h2[e * 40 + col2] : 0.f;
        float f1 = (col2 < 40 && e < 120) ? h2[(e + 1) * 40 + col2] : 0.f;
        b[j] = pkbf(f0, f1);
      }
      acc = wmma_bf16(a, b, acc);
    }
    int hi = lane >> 4;
    float* dump4 = hpre + tid;               // hpre dead during P4
#pragma unroll
    for (int r = 0; r < 8; ++r) {
      int c = r + hi * 8;
      float* dst = (c < 3 && col2 < 40) ? (Xs + col2 * 4 + c) : dump4;
      *dst = acc[r];                         // unconditional, addr-selected
    }
  }
  __syncthreads();

  // ---- P5: attention + output softmax (one lane per item) ---------------
  if (tid < 8 && (b0 + tid) < (long)nB) {
    const float* P = wsf;
    const float lam = P[0];
    float X[3][5];
#pragma unroll
    for (int c = 0; c < 3; ++c)
#pragma unroll
      for (int t = 0; t < 5; ++t) X[c][t] = Xs[(tid * 5 + t) * 4 + c];

    float y[3];
#pragma unroll
    for (int c = 0; c < 3; ++c) {
      float z[5], mx = -1e30f;
#pragma unroll
      for (int t2 = 0; t2 < 5; ++t2) {
        float s = 0.f;
#pragma unroll
        for (int t = 0; t < 5; ++t) s += X[c][t] * P[1 + t * 5 + t2];
        z[t2] = s;
        mx = fmaxf(mx, s);
      }
      float se = 0.f, aw[5];
#pragma unroll
      for (int t2 = 0; t2 < 5; ++t2) { aw[t2] = expf(z[t2] - mx); se += aw[t2]; }
      float inv = 1.f / se;
      float yv = 0.f;
#pragma unroll
      for (int t = 0; t < 5; ++t) {
        float xp = lam * X[c][t] + (1.f - lam) * X[c][t] * (aw[t] * inv);
        yv += xp * P[26 + t];
      }
      y[c] = yv + TB[c];
    }
    float mx = fmaxf(y[0], fmaxf(y[1], y[2]));
    float e0 = expf(y[0] - mx), e1 = expf(y[1] - mx), e2 = expf(y[2] - mx);
    float s = e0 + e1 + e2;
    float* o = out + (b0 + tid) * 3;
    o[0] = e0 / s; o[1] = e1 / s; o[2] = e2 / s;
  }
}

// ---------------------------------------------------------------------------
extern "C" void kernel_launch(void* const* d_in, const int* in_sizes, int n_in,
                              void* d_out, int out_size, void* d_ws, size_t ws_size,
                              hipStream_t stream) {
  const float* x   = (const float*)d_in[0];
  const float* W1  = (const float*)d_in[1];
  const float* W2  = (const float*)d_in[2];
  const float* BLB = (const float*)d_in[3];
  const float* TW1 = (const float*)d_in[4];
  const float* TW  = (const float*)d_in[5];
  const float* TW2 = (const float*)d_in[6];
  const float* TB  = (const float*)d_in[7];
  const float* l   = (const float*)d_in[8];
  float* out = (float*)d_out;
  float* wsf = (float*)d_ws;

  const int nB = in_sizes[0] / 400;          // batch (x is [B,40,10])

  m_btabl_prep<<<1, 256, 0, stream>>>(W1, W2, TW1, TW, TW2, l, wsf);

  const int nblk = (nB + 7) / 8;
  m_btabl_main<<<nblk, 256, 0, stream>>>(x, BLB, TB, wsf, out, nB);
}
